// Sreparameterize_6889127542973
// MI455X (gfx1250) — matrix operand order repaired and checked by reading, not compiled, not run on hardware
//
#include <hip/hip_runtime.h>
#include <hip/hip_bf16.h>

// ---------------- types for WMMA ----------------
typedef __attribute__((ext_vector_type(16))) __bf16 v16bf;
typedef __attribute__((ext_vector_type(8)))  float  v8f;

#define K_DIM          512
#define Z_DIM          64
#define NCOL           65      // 64 mu columns + 1 kappa column
#define TILES          5       // five 16-wide N tiles (cols 0..79; 65..79 dead)
#define ROWS_PER_BLOCK 128
#define THREADS        256
#define KHALF          256     // K staged in two halves of 256

union BFrag { uint4 q[2]; v16bf v; };

// ---------------- stateless counter RNG ----------------
__device__ __forceinline__ unsigned pcg_hash(unsigned v) {
    v = v * 747796405u + 2891336453u;
    unsigned x = ((v >> ((v >> 28u) + 4u)) ^ v) * 277803737u;
    return (x >> 22u) ^ x;
}
__device__ __forceinline__ float u01(unsigned row, unsigned stream, unsigned ctr) {
    unsigned h = pcg_hash(row * 0x9E3779B9u ^ (stream * 0x85EBCA6Bu + ctr * 0xC2B2AE35u + 0x165667B1u));
    return ((h >> 8) + 1u) * (1.0f / 16777216.0f);   // (0,1]
}
// deterministic normal #i for (row, stream) -- reproducible across passes
__device__ __forceinline__ float normal_at(unsigned row, unsigned stream, unsigned i) {
    float u1 = u01(row, stream, 2u * i);
    float u2 = u01(row, stream, 2u * i + 1u);
    return sqrtf(-2.0f * logf(u1)) * cosf(6.28318530718f * u2);
}
// Marsaglia-Tsang Gamma(alpha) sampler, alpha = d0 + 1/3
__device__ __forceinline__ float gamma_ss(unsigned row, unsigned stream, unsigned* ctr,
                                          float d0, float c0) {
    for (int it = 0; it < 32; ++it) {
        float u1 = u01(row, stream, (*ctr)++);
        float u2 = u01(row, stream, (*ctr)++);
        float xn = sqrtf(-2.0f * logf(u1)) * cosf(6.28318530718f * u2);
        float v = 1.0f + c0 * xn;
        if (v <= 0.0f) continue;
        v = v * v * v;
        float u = u01(row, stream, (*ctr)++);
        if (logf(u) < 0.5f * xn * xn + d0 + d0 * (logf(v) - v)) return d0 * v;
    }
    return d0;  // bounded loop for determinism
}

// ---------------- fused kernel ----------------
__global__ void __launch_bounds__(THREADS)
vmf_reparam_kernel(const float* __restrict__ x,
                   const float* __restrict__ mu_w,
                   const float* __restrict__ mu_b,
                   const float* __restrict__ k_w,
                   const float* __restrict__ k_b,
                   float* __restrict__ out) {
    extern __shared__ unsigned char smem[];
    unsigned short* Bs  = (unsigned short*)smem;   // [80 cols][256 k] bf16, 40 KB
    float*          Mrw = (float*)smem;            // reused after GEMM: 8*16*65 f32

    const int tid  = threadIdx.x;
    const int wave = tid >> 5;
    const int ln   = tid & 31;
    const int nloc = ln & 15;
    const int hsel = (ln < 16) ? 0 : 1;            // half-wave select
    const int rowA = blockIdx.x * ROWS_PER_BLOCK + wave * 16 + nloc;
    const float* __restrict__ xrow = x + (size_t)rowA * K_DIM;

    v8f acc[TILES];
#pragma unroll
    for (int t = 0; t < TILES; ++t)
#pragma unroll
        for (int j = 0; j < 8; ++j) acc[t][j] = 0.0f;

    // zero dead B' columns 65..79 once (never overwritten until Mrw reuse)
    for (int idx = tid; idx < 15 * KHALF; idx += THREADS)
        Bs[(65 + (idx >> 8)) * KHALF + (idx & 255)] = 0;

    // ---- A-load double buffer: issue K-step loads well before their use ----
    float4 n0, n1, n2, n3;
    const int aoff = (hsel ? 8 : 0);
    auto issueA = [&](int kh, int s) {
        const int kbase = kh * KHALF + s * 32 + aoff;
        n0 = *(const float4*)(xrow + kbase);
        n1 = *(const float4*)(xrow + kbase + 4);
        n2 = *(const float4*)(xrow + kbase + 16);
        n3 = *(const float4*)(xrow + kbase + 20);
        // pre-touch HBM lines ~4 K-steps ahead (speculative, OOB-safe)
        __builtin_prefetch(xrow + kbase + 128, 0, 0);
    };
    issueA(0, 0);

    // A fragment registers are double-buffered too: the cvt chain for step s+1
    // writes the *other* register set, killing the WMMA->VALU WAR hazard NOPs.
    v16bf afrag[2];

#pragma unroll
    for (int kh = 0; kh < 2; ++kh) {
        // ---- stage B' = [mu_w | k_w] for this K half, column-major bf16 ----
        // Bursted: 8 independent global loads in flight per wait, instead of a
        // serialized load->wait->store chain.
        {
            const int n4  = (tid & 15) * 4;                 // column group
            const int kk0 = tid >> 4;                       // base k row
#pragma unroll
            for (int b2 = 0; b2 < 2; ++b2) {
                float4 vv[8];
#pragma unroll
                for (int u = 0; u < 8; ++u) {
                    const int kk = kk0 + (b2 * 8 + u) * 16;
                    vv[u] = *(const float4*)(mu_w + (size_t)(kh * KHALF + kk) * Z_DIM + n4);
                }
#pragma unroll
                for (int u = 0; u < 8; ++u) {
                    const int kk = kk0 + (b2 * 8 + u) * 16;
                    __bf16 h0 = (__bf16)vv[u].x, h1 = (__bf16)vv[u].y;
                    __bf16 h2 = (__bf16)vv[u].z, h3 = (__bf16)vv[u].w;
                    Bs[(n4 + 0) * KHALF + kk] = __builtin_bit_cast(unsigned short, h0);
                    Bs[(n4 + 1) * KHALF + kk] = __builtin_bit_cast(unsigned short, h1);
                    Bs[(n4 + 2) * KHALF + kk] = __builtin_bit_cast(unsigned short, h2);
                    Bs[(n4 + 3) * KHALF + kk] = __builtin_bit_cast(unsigned short, h3);
                }
            }
            float kv = k_w[kh * KHALF + tid];
            __bf16 hk = (__bf16)kv;
            Bs[64 * KHALF + tid] = __builtin_bit_cast(unsigned short, hk);
        }
        __syncthreads();

        // ---- GEMM over this K half: 8 steps of K=32, software pipelined ----
#pragma unroll
        for (int s = 0; s < 8; ++s) {
            // convert the in-flight A chunk into this step's fragment buffer
            v16bf& a = afrag[s & 1];
            a[0]  = (__bf16)n0.x; a[1]  = (__bf16)n0.y; a[2]  = (__bf16)n0.z; a[3]  = (__bf16)n0.w;
            a[4]  = (__bf16)n1.x; a[5]  = (__bf16)n1.y; a[6]  = (__bf16)n1.z; a[7]  = (__bf16)n1.w;
            a[8]  = (__bf16)n2.x; a[9]  = (__bf16)n2.y; a[10] = (__bf16)n2.z; a[11] = (__bf16)n2.w;
            a[12] = (__bf16)n3.x; a[13] = (__bf16)n3.y; a[14] = (__bf16)n3.z; a[15] = (__bf16)n3.w;

            // issue next K-step's A loads (across the barrier for the last step)
            if (s < 7)           issueA(kh, s + 1);
            else if (kh == 0)    issueA(1, 0);

            // batch all 5 B fragments (10x ds_load_b128), then the WMMA chain
            const int k0 = s * 32 + (hsel ? 8 : 0);
            BFrag bf[TILES];
#pragma unroll
            for (int t = 0; t < TILES; ++t) {
                const uint4* bp = (const uint4*)(Bs + (t * 16 + nloc) * KHALF + k0);
                bf[t].q[0] = bp[0];
                bf[t].q[1] = bp[2];
            }
#pragma unroll
            for (int t = 0; t < TILES; ++t)
                acc[t] = __builtin_amdgcn_wmma_f32_16x16x32_bf16(
                    false, a, false, bf[t].v, (short)0, acc[t], false, false);
        }
        __syncthreads();
    }

    // ---- spill raw mu / kappa accumulators to LDS (region reused) ----
    {
        float* M = Mrw + wave * (16 * NCOL);
#pragma unroll
        for (int t = 0; t < 4; ++t)
#pragma unroll
            for (int j = 0; j < 8; ++j)
                M[(j + 8 * hsel) * NCOL + 16 * t + nloc] = acc[t][j];
        if (nloc == 0)
#pragma unroll
            for (int j = 0; j < 8; ++j)
                M[(j + 8 * hsel) * NCOL + 64] = acc[4][j];
    }
    __syncthreads();

    // ---- epilogue: one thread per row ----
    if (tid < ROWS_PER_BLOCK) {
        float* M = Mrw + (tid >> 4) * (16 * NCOL) + (tid & 15) * NCOL;
        const unsigned grow = blockIdx.x * ROWS_PER_BLOCK + tid;

        // bias + norm of mu
        float s2 = 0.0f;
        for (int i = 0; i < Z_DIM; ++i) {
            float m = M[i] + mu_b[i];
            M[i] = m;
            s2 += m * m;
        }
        float rnorm = rsqrtf(fmaxf(s2, 1e-30f));

        // kappa = softplus(x@k_w + k_b) + 1
        float kz = M[64] + k_b[0];
        float kp = (kz > 20.0f) ? kz : log1pf(expf(kz));
        float k  = kp + 1.0f;

        const float mm1 = 63.0f;                       // m - 1
        float c      = sqrtf(4.0f * k * k + mm1 * mm1);
        float b_true = (c - 2.0f * k) / mm1;
        float b_app  = mm1 / (4.0f * k);
        float sbl    = fminf(fmaxf(k - 10.0f, 0.0f), 1.0f);
        float b      = b_app * sbl + b_true * (1.0f - sbl);
        float a_     = (mm1 + 2.0f * k + c) * 0.25f;
        float d      = 4.0f * a_ * b / (1.0f + b) - mm1 * logf(mm1);

        // rejection-sample e via Beta(31.5, 31.5) = G1/(G1+G2)
        const float alpha = mm1 * 0.5f;
        const float d0 = alpha - (1.0f / 3.0f);
        const float c0 = rsqrtf(9.0f * d0);
        unsigned ctr = 0;
        float e = 0.5f;
        for (int it = 0; it < 64; ++it) {
            float g1 = gamma_ss(grow, 0u, &ctr, d0, c0);
            float g2 = gamma_ss(grow, 0u, &ctr, d0, c0);
            float e_ = g1 / (g1 + g2);
            float u  = u01(grow, 0u, ctr++);
            float t2 = 2.0f * a_ * b / (1.0f - (1.0f - b) * e_);
            if ((mm1 * logf(t2) - t2 + d) > logf(u)) { e = e_; break; }
        }
        float w  = (1.0f - (1.0f + b) * e) / (1.0f - (1.0f - b) * e);
        float w_ = sqrtf(fmaxf(1.0f - w * w, 1e-10f));

        // pass A: |v|^2, Householder vector norm, raw dot  (normals regenerated)
        float vs2 = 0.0f, hn2 = 0.0f, draw = 0.0f, hr0 = 0.0f;
        for (int i = 0; i < Z_DIM; ++i) {
            float mn = M[i] * rnorm;
            float hr = ((i == 0) ? 1.0f : 0.0f) - mn;   // e1 - mu_n
            hn2 += hr * hr;
            if (i == 0) {
                hr0 = hr;
            } else {
                float nv = normal_at(grow, 1u, (unsigned)(i - 1));
                vs2 += nv * nv;
                draw += nv * hr;
            }
        }
        float rv  = rsqrtf(fmaxf(vs2, 1e-30f));
        float rh  = rsqrtf(fmaxf(hn2, 1e-30f));
        float dot = (w * hr0 + w_ * rv * draw) * rh;

        // pass B: z = xs - 2*dot*u_h
        float* o = out + (size_t)grow * Z_DIM;
        for (int i = 0; i < Z_DIM; ++i) {
            float mn = M[i] * rnorm;
            float hr = ((i == 0) ? 1.0f : 0.0f) - mn;
            float xs = (i == 0) ? w : (w_ * rv * normal_at(grow, 1u, (unsigned)(i - 1)));
            o[i] = xs - 2.0f * dot * hr * rh;
        }
    }
}

extern "C" void kernel_launch(void* const* d_in, const int* in_sizes, int n_in,
                              void* d_out, int out_size, void* d_ws, size_t ws_size,
                              hipStream_t stream) {
    const float* x    = (const float*)d_in[0];
    const float* mu_w = (const float*)d_in[1];
    const float* mu_b = (const float*)d_in[2];
    const float* k_w  = (const float*)d_in[3];
    const float* k_b  = (const float*)d_in[4];
    float* out = (float*)d_out;

    const int B    = in_sizes[0] / K_DIM;           // 131072
    const int grid = B / ROWS_PER_BLOCK;            // 1024
    const size_t shmem = (size_t)(TILES * 16) * KHALF * sizeof(unsigned short); // 40 KB

    vmf_reparam_kernel<<<grid, THREADS, shmem, stream>>>(x, mu_w, mu_b, k_w, k_b, out);
}
